// MS_MSA_16896401342993
// MI455X (gfx1250) — compile-verified, matrix operand order checked by reading, not run on hardware
//
#include <hip/hip_runtime.h>
#include <hip/hip_bf16.h>
#include <math.h>

// ---------------------------------------------------------------------------
// MS-MSA for MI455X (gfx1250): bf16 WMMA GEMMs (async-LDS staged) + fused f32
// elementwise stages.
// ---------------------------------------------------------------------------

typedef __bf16 bf16;
typedef __attribute__((ext_vector_type(16))) __bf16 bf16x16;
typedef __attribute__((ext_vector_type(8)))  float  f32x8;
typedef __attribute__((ext_vector_type(4)))  unsigned int u32x4;

#define C_     256
#define B_     4
#define HH     128
#define WW     128
#define NPIX   16384        // H*W
#define MROWS  65536        // b * NPIX
#define HEADS_ 8
#define DHEAD  32

// ======================= f32 -> bf16 conversion ============================
__global__ void cvt_bf16_kernel(const float* __restrict__ src,
                                bf16* __restrict__ dst, int count) {
  int i = blockIdx.x * blockDim.x + threadIdx.x;
  if (i < count) dst[i] = (bf16)src[i];
}

// ============ f32 [K,N] -> bf16 transposed [N,K] (weights only) ============
__global__ void cvt_bf16_T_kernel(const float* __restrict__ src,
                                  bf16* __restrict__ dst) {
  int i = blockIdx.x * blockDim.x + threadIdx.x;   // < 65536
  int k = i & 255;
  int n = i >> 8;
  dst[n * 256 + k] = (bf16)src[k * 256 + n];
}

// ======================= WMMA GEMM (bf16 x bf16 -> f32) ====================
// C[M,N] = A[M,K] @ Bt[N,K]^T (+ bias[N]); optional bf16 copy of C.
// Block: 256 threads = 8 waves (wave32). Block tile 128x64, wave tile 32x32,
// K-step 64 => 8 v_wmma per barrier pair. A/B tiles staged with
// GLOBAL_LOAD_ASYNC_TO_LDS_B128 (ASYNCcnt), fenced with s_wait_asynccnt.
#define BM 128
#define BN 64
#define BK 64

__device__ __forceinline__ void async_ld16(const bf16* gp, const bf16* lp) {
  unsigned lds = (unsigned)(unsigned long long)lp;
  asm volatile("global_load_async_to_lds_b128 %0, %1, off"
               :: "v"(lds), "v"((unsigned long long)gp)
               : "memory");
}

__device__ __forceinline__ bf16x16 lds_frag(const bf16* p) {
  union { bf16x16 v; u32x4 u[2]; } f;
  f.u[0] = *(const u32x4*)p;
  f.u[1] = *(const u32x4*)(p + 16);
  return f.v;
}

__global__ __launch_bounds__(256) void gemm_wmma_kernel(
    const bf16* __restrict__ A, const bf16* __restrict__ Bt,
    const float* __restrict__ bias, float* __restrict__ Cf,
    bf16* __restrict__ Cbf, int M, int N, int K)
{
  __shared__ __align__(16) bf16 lA[BM * BK];       // [m][k]
  __shared__ __align__(16) bf16 lB[BN * BK];       // [n][k]

  const int tid  = threadIdx.x;
  const int lane = tid & 31;
  const int wave = tid >> 5;        // 0..7
  const int wm   = wave & 3;        // 0..3 -> 32-row slice
  const int wn   = wave >> 2;       // 0..1 -> 32-col slice
  const int half = lane >> 4;       // 0/1
  const int l16  = lane & 15;

  const int rowBase = blockIdx.y * BM;
  const int colBase = blockIdx.x * BN;

  f32x8 acc00 = {}, acc01 = {}, acc10 = {}, acc11 = {};

  for (int k0 = 0; k0 < K; k0 += BK) {
    // ---- async stage A tile: 128x64 bf16 = 1024 16B chunks, 4/thread ----
#pragma unroll
    for (int j = 0; j < 4; ++j) {
      int c  = tid + j * 256;
      int r  = c >> 3;
      int c8 = (c & 7) << 3;
      async_ld16(A + (size_t)(rowBase + r) * K + k0 + c8, &lA[r * BK + c8]);
    }
    // ---- async stage B tile: 64x64 bf16 = 512 16B chunks, 2/thread ----
#pragma unroll
    for (int j = 0; j < 2; ++j) {
      int c  = tid + j * 256;
      int r  = c >> 3;
      int c8 = (c & 7) << 3;
      async_ld16(Bt + (size_t)(colBase + r) * K + k0 + c8, &lB[r * BK + c8]);
    }
    // prefetch next A tile into the cache hierarchy
    if (k0 + BK < K) {
      int r = tid >> 1;
      __builtin_prefetch(A + (size_t)(rowBase + r) * K + (k0 + BK), 0, 0);
    }
    asm volatile("s_wait_asynccnt 0x0" ::: "memory");
    __syncthreads();

    // ---- compute: 2x2 wave tile, two k-halves -> 8 wmma ----
#pragma unroll
    for (int ka = 0; ka < BK; ka += 32) {
      const bf16* pa0 = &lA[(wm * 32 + 0  + l16) * BK + ka + half * 8];
      const bf16* pa1 = &lA[(wm * 32 + 16 + l16) * BK + ka + half * 8];
      const bf16* pb0 = &lB[(wn * 32 + 0  + l16) * BK + ka + half * 8];
      const bf16* pb1 = &lB[(wn * 32 + 16 + l16) * BK + ka + half * 8];
      bf16x16 fa0 = lds_frag(pa0);
      bf16x16 fa1 = lds_frag(pa1);
      bf16x16 fb0 = lds_frag(pb0);
      bf16x16 fb1 = lds_frag(pb1);
      acc00 = __builtin_amdgcn_wmma_f32_16x16x32_bf16(false, fa0, false, fb0,
                                                      (short)0, acc00, false, false);
      acc01 = __builtin_amdgcn_wmma_f32_16x16x32_bf16(false, fa0, false, fb1,
                                                      (short)0, acc01, false, false);
      acc10 = __builtin_amdgcn_wmma_f32_16x16x32_bf16(false, fa1, false, fb0,
                                                      (short)0, acc10, false, false);
      acc11 = __builtin_amdgcn_wmma_f32_16x16x32_bf16(false, fa1, false, fb1,
                                                      (short)0, acc11, false, false);
    }
    __syncthreads();
  }

  // ---- epilogue: D layout: VGPR r -> M = r + 8*half, N = l16 ----
  const int n0 = colBase + wn * 32 + l16;
  const float bv0 = bias ? bias[n0]      : 0.0f;
  const float bv1 = bias ? bias[n0 + 16] : 0.0f;
  const int m0 = rowBase + wm * 32 + half * 8;
#pragma unroll
  for (int r = 0; r < 8; ++r) {
    float v00 = acc00[r] + bv0;
    float v01 = acc01[r] + bv1;
    float v10 = acc10[r] + bv0;
    float v11 = acc11[r] + bv1;
    size_t ra = (size_t)(m0 + r) * N;
    size_t rb = (size_t)(m0 + 16 + r) * N;
    Cf[ra + n0]      = v00;
    Cf[ra + n0 + 16] = v01;
    Cf[rb + n0]      = v10;
    Cf[rb + n0 + 16] = v11;
    if (Cbf) {
      Cbf[ra + n0]      = (bf16)v00;
      Cbf[ra + n0 + 16] = (bf16)v01;
      Cbf[rb + n0]      = (bf16)v10;
      Cbf[rb + n0 + 16] = (bf16)v11;
    }
  }
}

// ============== fused mask mechanism: dw5x5 + sigmoid + gating =============
__global__ __launch_bounds__(256) void mask_mech_kernel(
    const float* __restrict__ m, const float* __restrict__ t2,
    const float* __restrict__ v, const float* __restrict__ dww,
    const float* __restrict__ dwb, float* __restrict__ vg)
{
  int i  = blockIdx.x * blockDim.x + threadIdx.x;   // < 2^24
  int ch = i & 255;
  int x  = (i >> 8) & 127;
  int y  = (i >> 15) & 127;
  int b  = i >> 22;
  float accum = dwb[ch];
#pragma unroll
  for (int dy = 0; dy < 5; ++dy) {
    int yy = y + dy - 2;
    if ((unsigned)yy > 127u) continue;
#pragma unroll
    for (int dx = 0; dx < 5; ++dx) {
      int xx = x + dx - 2;
      if ((unsigned)xx > 127u) continue;
      size_t idx = ((((size_t)b * 128 + yy) * 128) + xx) * 256 + ch;
      accum += t2[idx] * dww[(dy * 5 + dx) * 256 + ch];
    }
  }
  float amap = 1.0f / (1.0f + expf(-accum));
  float mv   = m[i];
  vg[i] = v[i] * (mv * amap + mv);
}

// ================= L2 row norms over n (per b, per channel) ================
__global__ __launch_bounds__(256) void rownorm_kernel(
    const float* __restrict__ t, float* __restrict__ norms)
{
  __shared__ float red[256];
  int b  = blockIdx.x >> 8;
  int cc = blockIdx.x & 255;
  const float* base = t + (size_t)b * NPIX * C_ + cc;
  float s = 0.f;
  for (int nn = threadIdx.x; nn < NPIX; nn += 256) {
    float x = base[(size_t)nn * C_];
    s += x * x;
  }
  red[threadIdx.x] = s;
  __syncthreads();
  for (int off = 128; off > 0; off >>= 1) {
    if (threadIdx.x < (unsigned)off) red[threadIdx.x] += red[threadIdx.x + off];
    __syncthreads();
  }
  if (threadIdx.x == 0) norms[blockIdx.x] = sqrtf(red[0]);
}

// ========== Gram matrix G[b,h,d,e] = sum_n k[.,d]*q[.,e]  (32x32) ==========
__global__ __launch_bounds__(256) void attn_gram_kernel(
    const float* __restrict__ k, const float* __restrict__ q,
    float* __restrict__ G)
{
  __shared__ float lK[64][32];
  __shared__ float lQ[64][32];
  int bh = blockIdx.x;                 // 0..31
  int b = bh >> 3, h = bh & 7;
  const float* kb = k + (size_t)b * NPIX * C_ + h * DHEAD;
  const float* qb = q + (size_t)b * NPIX * C_ + h * DHEAD;
  int tid = threadIdx.x;
  int e  = tid & 31;
  int d0 = (tid >> 5) << 2;
  float a0 = 0, a1 = 0, a2 = 0, a3 = 0;
  for (int n0 = 0; n0 < NPIX; n0 += 64) {
    __syncthreads();
    for (int j = 0; j < 8; ++j) {
      int idx = tid + j * 256;
      int nn = idx >> 5, d = idx & 31;
      lK[nn][d] = kb[(size_t)(n0 + nn) * C_ + d];
      lQ[nn][d] = qb[(size_t)(n0 + nn) * C_ + d];
    }
    __syncthreads();
    for (int nn = 0; nn < 64; ++nn) {
      float qv = lQ[nn][e];
      a0 += lK[nn][d0 + 0] * qv;
      a1 += lK[nn][d0 + 1] * qv;
      a2 += lK[nn][d0 + 2] * qv;
      a3 += lK[nn][d0 + 3] * qv;
    }
  }
  float* Gb = G + (size_t)bh * 1024;
  Gb[(d0 + 0) * 32 + e] = a0;
  Gb[(d0 + 1) * 32 + e] = a1;
  Gb[(d0 + 2) * 32 + e] = a2;
  Gb[(d0 + 3) * 32 + e] = a3;
}

// =========== fold l2norm scales + rescale, softmax over e (rows) ===========
__global__ void softmax_kernel(float* __restrict__ G, const float* __restrict__ kn,
                               const float* __restrict__ qn,
                               const float* __restrict__ rescale)
{
  int bh = blockIdx.x;
  int b = bh >> 3, h = bh & 7;
  int d = threadIdx.x;                 // 0..31, one row per lane
  float* Gb = G + (size_t)bh * 1024 + d * 32;
  const float* knb = kn + b * 256 + h * 32;
  const float* qnb = qn + b * 256 + h * 32;
  float kd = fmaxf(knb[d], 1e-6f);
  float rs = rescale[h];
  float vals[32];
  float mx = -1e30f;
  for (int e = 0; e < 32; ++e) {
    float v = Gb[e] / (kd * fmaxf(qnb[e], 1e-6f)) * rs;
    vals[e] = v;
    mx = fmaxf(mx, v);
  }
  float s = 0.f;
  for (int e = 0; e < 32; ++e) { vals[e] = expf(vals[e] - mx); s += vals[e]; }
  float inv = 1.0f / s;
  for (int e = 0; e < 32; ++e) Gb[e] = vals[e] * inv;
}

// === attention output: ao[b,n,h*32+d] = sum_e attn[d,e]*vg[b,n,h*32+e] =====
__global__ __launch_bounds__(256) void attn_out_kernel(
    const float* __restrict__ attn, const float* __restrict__ vg,
    bf16* __restrict__ ao_bf)
{
  int i    = blockIdx.x * blockDim.x + threadIdx.x;  // 16.7M
  int cidx = i & 255;
  int d = cidx & 31;
  int h = cidx >> 5;
  int b = i >> 22;
  const float* arow = attn + (((size_t)b * 8 + h) * 32 + d) * 32;
  const float* vrow = vg + (size_t)(i - d);          // e=0 base for same (b,n,h)
  float acc = 0.f;
#pragma unroll
  for (int e = 0; e < 32; ++e) acc += arow[e] * vrow[e];
  ao_bf[i] = (bf16)acc;
}

// ============ positional embedding: t = gelu(dwconv3x3(v, pe1)) ============
__global__ __launch_bounds__(256) void pe1_kernel(
    const float* __restrict__ v, const float* __restrict__ w,
    float* __restrict__ t)
{
  int i  = blockIdx.x * blockDim.x + threadIdx.x;
  int ch = i & 255;
  int x  = (i >> 8) & 127;
  int y  = (i >> 15) & 127;
  int b  = i >> 22;
  float acc = 0.f;
#pragma unroll
  for (int dy = 0; dy < 3; ++dy) {
    int yy = y + dy - 1;
    if ((unsigned)yy > 127u) continue;
#pragma unroll
    for (int dx = 0; dx < 3; ++dx) {
      int xx = x + dx - 1;
      if ((unsigned)xx > 127u) continue;
      acc += v[((((size_t)b * 128 + yy) * 128) + xx) * 256 + ch] *
             w[(dy * 3 + dx) * 256 + ch];
    }
  }
  t[i] = 0.5f * acc * (1.0f + erff(acc * 0.70710678118f));  // exact GELU
}

// ============ out += dwconv3x3(t, pe2)  (out already holds proj) ===========
__global__ __launch_bounds__(256) void pe2_kernel(
    const float* __restrict__ t, const float* __restrict__ w,
    float* __restrict__ out)
{
  int i  = blockIdx.x * blockDim.x + threadIdx.x;
  int ch = i & 255;
  int x  = (i >> 8) & 127;
  int y  = (i >> 15) & 127;
  int b  = i >> 22;
  float acc = 0.f;
#pragma unroll
  for (int dy = 0; dy < 3; ++dy) {
    int yy = y + dy - 1;
    if ((unsigned)yy > 127u) continue;
#pragma unroll
    for (int dx = 0; dx < 3; ++dx) {
      int xx = x + dx - 1;
      if ((unsigned)xx > 127u) continue;
      acc += t[((((size_t)b * 128 + yy) * 128) + xx) * 256 + ch] *
             w[(dy * 3 + dx) * 256 + ch];
    }
  }
  out[i] += acc;
}

// ===========================================================================
extern "C" void kernel_launch(void* const* d_in, const int* in_sizes, int n_in,
                              void* d_out, int out_size, void* d_ws, size_t ws_size,
                              hipStream_t stream)
{
  (void)in_sizes; (void)n_in; (void)out_size; (void)ws_size;

  const float* x_in    = (const float*)d_in[0];
  const float* mask    = (const float*)d_in[1];
  const float* Wq      = (const float*)d_in[2];
  const float* Wk      = (const float*)d_in[3];
  const float* Wv      = (const float*)d_in[4];
  const float* rescale = (const float*)d_in[5];
  const float* Wproj   = (const float*)d_in[6];
  const float* bproj   = (const float*)d_in[7];
  const float* c1w     = (const float*)d_in[8];
  const float* c1b     = (const float*)d_in[9];
  const float* c2w     = (const float*)d_in[10];
  const float* c2b     = (const float*)d_in[11];
  const float* dww     = (const float*)d_in[12];
  const float* dwb     = (const float*)d_in[13];
  const float* pe1w    = (const float*)d_in[14];
  const float* pe2w    = (const float*)d_in[15];
  float* out = (float*)d_out;

  const size_t NE = (size_t)MROWS * C_;   // 16,777,216 elements per tensor
  const size_t WE = (size_t)C_ * C_;      // 65,536 per weight matrix

  char* ws = (char*)d_ws;
  size_t off = 0;
  bf16* x_bf    = (bf16*)(ws + off); off += NE * 2;   // later reused as ao_bf
  bf16* mask_bf = (bf16*)(ws + off); off += NE * 2;
  bf16* m_bf    = (bf16*)(ws + off); off += NE * 2;
  bf16* wq_bf   = (bf16*)(ws + off); off += WE * 2;   // all weights stored [N][K]
  bf16* wk_bf   = (bf16*)(ws + off); off += WE * 2;
  bf16* wv_bf   = (bf16*)(ws + off); off += WE * 2;
  bf16* wp_bf   = (bf16*)(ws + off); off += WE * 2;
  bf16* c1_bf   = (bf16*)(ws + off); off += WE * 2;
  bf16* c2_bf   = (bf16*)(ws + off); off += WE * 2;
  float* qn     = (float*)(ws + off); off += 1024 * 4;
  float* kn     = (float*)(ws + off); off += 1024 * 4;
  float* G      = (float*)(ws + off); off += (size_t)B_ * HEADS_ * 32 * 32 * 4;
  off = (off + 255) & ~(size_t)255;
  float* q      = (float*)(ws + off); off += NE * 4;  // later reused as pe temp
  float* k      = (float*)(ws + off); off += NE * 4;
  float* v      = (float*)(ws + off); off += NE * 4;
  float* m      = (float*)(ws + off); off += NE * 4;  // vg written in place
  float* t2     = (float*)(ws + off); off += NE * 4;
  bf16*  ao_bf  = x_bf;     // alias: x_bf dead after QKV GEMMs
  float* pe_t   = q;        // alias: q dead after gram/norm stage
  float* vg     = m;        // in-place gating

  const dim3 blk(256);
  const dim3 gBig((unsigned)(NE / 256));              // 65536 blocks
  const dim3 gW((unsigned)(WE / 256));                // 256 blocks
  const dim3 gGemm(C_ / BN, MROWS / BM);              // (4, 512)

  // 1) precision conversion (weights transposed to [N][K] for async staging)
  cvt_bf16_T_kernel<<<gW,   blk, 0, stream>>>(Wq,    wq_bf);
  cvt_bf16_T_kernel<<<gW,   blk, 0, stream>>>(Wk,    wk_bf);
  cvt_bf16_T_kernel<<<gW,   blk, 0, stream>>>(Wv,    wv_bf);
  cvt_bf16_T_kernel<<<gW,   blk, 0, stream>>>(Wproj, wp_bf);
  cvt_bf16_T_kernel<<<gW,   blk, 0, stream>>>(c1w,   c1_bf);
  cvt_bf16_T_kernel<<<gW,   blk, 0, stream>>>(c2w,   c2_bf);
  cvt_bf16_kernel<<<gBig, blk, 0, stream>>>(x_in, x_bf,    (int)NE);
  cvt_bf16_kernel<<<gBig, blk, 0, stream>>>(mask, mask_bf, (int)NE);

  // 2) big WMMA GEMMs
  gemm_wmma_kernel<<<gGemm, blk, 0, stream>>>(x_bf,    wq_bf, nullptr, q,  nullptr, MROWS, C_, C_);
  gemm_wmma_kernel<<<gGemm, blk, 0, stream>>>(x_bf,    wk_bf, nullptr, k,  nullptr, MROWS, C_, C_);
  gemm_wmma_kernel<<<gGemm, blk, 0, stream>>>(x_bf,    wv_bf, nullptr, v,  nullptr, MROWS, C_, C_);
  gemm_wmma_kernel<<<gGemm, blk, 0, stream>>>(mask_bf, c1_bf, c1b,     m,  m_bf,    MROWS, C_, C_);
  gemm_wmma_kernel<<<gGemm, blk, 0, stream>>>(m_bf,    c2_bf, c2b,     t2, nullptr, MROWS, C_, C_);

  // 3) fused mask mechanism -> gated V (in place over m)
  mask_mech_kernel<<<gBig, blk, 0, stream>>>(m, t2, v, dww, dwb, vg);

  // 4) channel attention (small): norms, gram, softmax, apply
  rownorm_kernel<<<dim3(1024), blk, 0, stream>>>(q, qn);
  rownorm_kernel<<<dim3(1024), blk, 0, stream>>>(k, kn);
  attn_gram_kernel<<<dim3(32), blk, 0, stream>>>(k, q, G);
  softmax_kernel<<<dim3(32), dim3(32), 0, stream>>>(G, kn, qn, rescale);
  attn_out_kernel<<<gBig, blk, 0, stream>>>(G, vg, ao_bf);

  // 5) output projection straight into d_out (WMMA)
  gemm_wmma_kernel<<<gGemm, blk, 0, stream>>>(ao_bf, wp_bf, bproj, out, nullptr, MROWS, C_, C_);

  // 6) conv positional embedding on ungated v, accumulated into d_out
  pe1_kernel<<<gBig, blk, 0, stream>>>(v, pe1w, pe_t);
  pe2_kernel<<<gBig, blk, 0, stream>>>(pe_t, pe2w, out);
}
